// PredictionNetwork_72610717106540
// MI455X (gfx1250) — compile-verified, matrix operand order checked by reading, not run on hardware
//
#include <hip/hip_runtime.h>
#include <hip/hip_bf16.h>

// ---------------------------------------------------------------------------
// GVP-GNN forward for MI455X (gfx1250, wave32, WMMA).
// All dense contractions run through v_wmma_f32_16x16x32_bf16 with f32
// accumulation; norms / LayerNorm / activations in f32 VALU; segment sums via
// global_atomic_add_f32 into L2. The edge-message kernel processes MT=2
// 16-edge tiles per wave so every packed-B fragment load is amortized over
// two WMMAs (halves L2 weight traffic -> matrix-op bound).
// ---------------------------------------------------------------------------

#define DEVFN __device__ __forceinline__

typedef __attribute__((ext_vector_type(16))) __bf16 v16bf;
typedef __attribute__((ext_vector_type(8)))  float  v8f;

#define NNODE  4096
#define NEDGE  122880
#define NGRAPH 16
#define LRES   256

// Wave-private LDS ordering: DS ops are in-order per wave; just drain DScnt
// and stop the compiler from reordering around it.
#define LDSFENCE() asm volatile("s_wait_dscnt 0x0" ::: "memory")

DEVFN v8f vzero(){ v8f a;
#pragma unroll
  for(int r=0;r<8;r++) a[r]=0.f;
  return a; }

DEVFN v8f wmma_bf16(v16bf a, v16bf b, v8f c){
  return __builtin_amdgcn_wmma_f32_16x16x32_bf16(false,a,false,b,(short)0,c,false,false);
}

// A-fragment layout (16x32 bf16): lane = 16*((k>>3)&1) + m ; per-lane element
// j = ((k>>4)&1)*8 + (k&7). Stored as [ktile][lane][16] -> contiguous 32B/lane.
DEVFN int aidx(int m,int k){
  int kt=k>>5, kk=k&31;
  int lane=(((kk>>3)&1)<<4)|m;
  int j=(((kk>>4)&1)<<3)|(kk&7);
  return kt*512 + lane*16 + j;
}
DEVFN void stA(__bf16* a,int m,int k,float v){ a[aidx(m,k)]=(__bf16)v; }

DEVFN v16bf ldA(const __bf16* a,int kt,int lane){
  return *(const v16bf*)(a + kt*512 + lane*16);
}
template<int NT>
DEVFN v16bf ldBp(const __bf16* Wp,int kt,int nt,int lane){
  return *(const v16bf*)(Wp + (size_t)(kt*NT+nt)*512 + lane*16);
}

// Pack f32 weight W[N][K] (row-major) into bf16 B-fragments (B = W^T), zero pad.
__global__ void pack_b_kernel(const float* __restrict__ W, __bf16* __restrict__ out,
                              int N, int K){
  int KT=(K+31)/32, NT=(N+15)/16;
  int total=KT*NT*512;
  for(int i=blockIdx.x*blockDim.x+threadIdx.x; i<total; i+=gridDim.x*blockDim.x){
    int j=i&15, lane=(i>>4)&31, frag=i>>9;
    int nt=frag%NT, kt=frag/NT;
    int n=nt*16+(lane&15);
    int k=kt*32 + ((j>>3)<<4) + ((lane>>4)<<3) + (j&7);
    float v=(n<N && k<K) ? W[(size_t)n*K+k] : 0.f;
    out[i]=(__bf16)v;
  }
}

// ---------------------------------------------------------------------------
// One GVP on MT 16-row tiles, executed by one wave. B fragments are loaded
// once per (kt,nt) and reused across the MT A-tiles.
//   aS/aV/aSout: staging regions; per-tile element strides asStride/avStride/
//                aoStride (0 when MT==1).
//   sOut[nt*MT+mt], vOut[(c*VOT+nt)*MT+mt]: returned C fragments.
// ---------------------------------------------------------------------------
template<int SI,int VI,int H,int SO,int VO,int ACT,int MT>
DEVFN void gvp_tile(__bf16* aS, __bf16* aV, __bf16* aSout, v8f* sOut, v8f* vOut,
                    int lane, const __bf16* whP, const __bf16* wsP,
                    const float* wsb, const __bf16* wvP,
                    int asStride=0, int avStride=0, int aoStride=0){
  constexpr int HT=(H+15)/16, VIKT=(VI+31)/32, SKT=(SI+H+31)/32, SOT=(SO+15)/16;
  constexpr int VOT=(VO+15)/16, HKT=(H+31)/32;
  // vh = V @ wh^T   (per coordinate)
  v8f vh[MT][3][HT];
#pragma unroll
  for(int c=0;c<3;c++){
#pragma unroll
    for(int nt=0;nt<HT;nt++){
#pragma unroll
      for(int mt=0;mt<MT;mt++) vh[mt][c][nt]=vzero();
#pragma unroll
      for(int kt=0;kt<VIKT;kt++){
        v16bf b=ldBp<HT>(whP,kt,nt,lane);
#pragma unroll
        for(int mt=0;mt<MT;mt++)
          vh[mt][c][nt]=wmma_bf16(ldA(aV+mt*avStride+c*1024,kt,lane),b,vh[mt][c][nt]);
      }
    }
  }
  // vn = ||vh|| elementwise over the 3 aligned C-fragments -> append to aS
#pragma unroll
  for(int mt=0;mt<MT;mt++){
#pragma unroll
    for(int nt=0;nt<HT;nt++){
      int col=nt*16+(lane&15);
      if(col<H){
        int mh=(lane>>4)*8;
#pragma unroll
        for(int r=0;r<8;r++){
          float s2=vh[mt][0][nt][r]*vh[mt][0][nt][r]
                  +vh[mt][1][nt][r]*vh[mt][1][nt][r]
                  +vh[mt][2][nt][r]*vh[mt][2][nt][r];
          stA(aS+mt*asStride, r+mh, SI+col, sqrtf(fmaxf(s2,1e-8f)));
        }
      }
    }
  }
  LDSFENCE();
  // s_out = [s, vn] @ ws^T + b  (optionally ReLU)
#pragma unroll
  for(int nt=0;nt<SOT;nt++){
    v8f acc[MT];
#pragma unroll
    for(int mt=0;mt<MT;mt++) acc[mt]=vzero();
#pragma unroll
    for(int kt=0;kt<SKT;kt++){
      v16bf b=ldBp<SOT>(wsP,kt,nt,lane);
#pragma unroll
      for(int mt=0;mt<MT;mt++)
        acc[mt]=wmma_bf16(ldA(aS+mt*asStride,kt,lane),b,acc[mt]);
    }
    int col=nt*16+(lane&15);
    float bias=(col<SO)?wsb[col]:0.f;
#pragma unroll
    for(int mt=0;mt<MT;mt++){
#pragma unroll
      for(int r=0;r<8;r++){ float x=acc[mt][r]+bias; if(ACT) x=fmaxf(x,0.f); acc[mt][r]=x; }
      if(aSout){
        if(col<SO){
          int mh=(lane>>4)*8;
#pragma unroll
          for(int r=0;r<8;r++) stA(aSout+mt*aoStride,r+mh,col,acc[mt][r]);
        }
      } else sOut[nt*MT+mt]=acc[mt];
    }
  }
  if constexpr (VO>0){
    // restage vh into aV and apply wv
#pragma unroll
    for(int mt=0;mt<MT;mt++)
      for(int i=lane;i<3072;i+=32) aV[mt*avStride+i]=(__bf16)0.f;
#pragma unroll
    for(int mt=0;mt<MT;mt++){
#pragma unroll
      for(int c=0;c<3;c++){
#pragma unroll
        for(int nt=0;nt<HT;nt++){
          int col=nt*16+(lane&15);
          if(col<H){
            int mh=(lane>>4)*8;
#pragma unroll
            for(int r=0;r<8;r++) stA(aV+mt*avStride+c*1024,r+mh,col,vh[mt][c][nt][r]);
          }
        }
      }
    }
    LDSFENCE();
#pragma unroll
    for(int c=0;c<3;c++){
#pragma unroll
      for(int nt=0;nt<VOT;nt++){
        v8f acc[MT];
#pragma unroll
        for(int mt=0;mt<MT;mt++) acc[mt]=vzero();
#pragma unroll
        for(int kt=0;kt<HKT;kt++){
          v16bf b=ldBp<VOT>(wvP,kt,nt,lane);
#pragma unroll
          for(int mt=0;mt<MT;mt++)
            acc[mt]=wmma_bf16(ldA(aV+mt*avStride+c*1024,kt,lane),b,acc[mt]);
        }
#pragma unroll
        for(int mt=0;mt<MT;mt++) vOut[(c*VOT+nt)*MT+mt]=acc[mt];
      }
    }
    if(ACT){ // vector_act = sigmoid(||v_out||) scaling
#pragma unroll
      for(int nt=0;nt<VOT;nt++){
#pragma unroll
        for(int mt=0;mt<MT;mt++){
#pragma unroll
          for(int r=0;r<8;r++){
            float x0=vOut[(0*VOT+nt)*MT+mt][r];
            float x1=vOut[(1*VOT+nt)*MT+mt][r];
            float x2=vOut[(2*VOT+nt)*MT+mt][r];
            float nrm=sqrtf(fmaxf(x0*x0+x1*x1+x2*x2,1e-8f));
            float sg=1.f/(1.f+__expf(-nrm));
            vOut[(0*VOT+nt)*MT+mt][r]=x0*sg;
            vOut[(1*VOT+nt)*MT+mt][r]=x1*sg;
            vOut[(2*VOT+nt)*MT+mt][r]=x2*sg;
          }
        }
      }
    }
  }
}

// ---------------------------------------------------------------------------
// Embedding kernels (tiny, per-element f32)
// ---------------------------------------------------------------------------
__global__ void embed_node_kernel(const float* __restrict__ ns, const float* __restrict__ nv,
  const float* lnb,const float* lnw,const float* wh,const float* wsb,const float* wsw,
  const float* wv, float* __restrict__ sN, float* __restrict__ vN){
  int n=blockIdx.x*blockDim.x+threadIdx.x;
  if(n>=NNODE) return;
  float s[6]; float mu=0.f;
#pragma unroll
  for(int k=0;k<6;k++){ s[k]=ns[n*6+k]; mu+=s[k]; }
  mu*=(1.f/6.f);
  float var=0.f;
#pragma unroll
  for(int k=0;k<6;k++){ float d=s[k]-mu; var+=d*d; }
  var*=(1.f/6.f);
  float rs=rsqrtf(var+1e-5f);
#pragma unroll
  for(int k=0;k<6;k++) s[k]=(s[k]-mu)*rs*lnw[k]+lnb[k];
  float v[3][3]; float acc=0.f;
#pragma unroll
  for(int i=0;i<3;i++){ float s2=0.f;
#pragma unroll
    for(int c=0;c<3;c++){ float x=nv[((size_t)n*3+i)*3+c]; v[i][c]=x; s2+=x*x; }
    acc+=fmaxf(s2,1e-8f); }
  float sc=rsqrtf(acc*(1.f/3.f));
#pragma unroll
  for(int i=0;i<3;i++) for(int c=0;c<3;c++) v[i][c]*=sc;
  float vh[16][3];
#pragma unroll
  for(int h=0;h<16;h++)
#pragma unroll
    for(int c=0;c<3;c++){
      float x=0.f;
#pragma unroll
      for(int i=0;i<3;i++) x+=wh[h*3+i]*v[i][c];
      vh[h][c]=x;
    }
  float vn[16];
#pragma unroll
  for(int h=0;h<16;h++)
    vn[h]=sqrtf(fmaxf(vh[h][0]*vh[h][0]+vh[h][1]*vh[h][1]+vh[h][2]*vh[h][2],1e-8f));
  for(int o=0;o<100;o++){
    float x=wsb[o];
#pragma unroll
    for(int k=0;k<6;k++) x+=wsw[o*22+k]*s[k];
#pragma unroll
    for(int h=0;h<16;h++) x+=wsw[o*22+6+h]*vn[h];
    sN[(size_t)n*100+o]=x;
  }
  for(int o=0;o<16;o++)
    for(int c=0;c<3;c++){
      float x=0.f;
#pragma unroll
      for(int h=0;h<16;h++) x+=wv[o*16+h]*vh[h][c];
      vN[((size_t)n*16+o)*3+c]=x;
    }
}

__global__ void embed_edge_kernel(const float* __restrict__ es, const float* __restrict__ ev,
  const float* lnb,const float* lnw,const float* wh,const float* wsb,const float* wsw,
  const float* wv, float* __restrict__ esE, float* __restrict__ evE){
  int e=blockIdx.x*blockDim.x+threadIdx.x;
  if(e>=NEDGE) return;
  float s[32]; float mu=0.f;
#pragma unroll
  for(int k=0;k<32;k++){ s[k]=es[(size_t)e*32+k]; mu+=s[k]; }
  mu*=(1.f/32.f);
  float var=0.f;
#pragma unroll
  for(int k=0;k<32;k++){ float d=s[k]-mu; var+=d*d; }
  var*=(1.f/32.f);
  float rs=rsqrtf(var+1e-5f);
#pragma unroll
  for(int k=0;k<32;k++) s[k]=(s[k]-mu)*rs*lnw[k]+lnb[k];
  float v0=ev[(size_t)e*3+0], v1=ev[(size_t)e*3+1], v2=ev[(size_t)e*3+2];
  float s2=fmaxf(v0*v0+v1*v1+v2*v2,1e-8f);
  float sc=rsqrtf(s2); v0*=sc; v1*=sc; v2*=sc;
  float w0=wh[0];
  float h0=w0*v0, h1=w0*v1, h2=w0*v2;
  float vn=sqrtf(fmaxf(h0*h0+h1*h1+h2*h2,1e-8f));
  for(int o=0;o<32;o++){
    float x=wsb[o]+wsw[o*33+32]*vn;
#pragma unroll
    for(int k=0;k<32;k++) x+=wsw[o*33+k]*s[k];
    esE[(size_t)e*32+o]=x;
  }
  float wv0=wv[0];
  evE[(size_t)e*3+0]=wv0*h0; evE[(size_t)e*3+1]=wv0*h1; evE[(size_t)e*3+2]=wv0*h2;
}

__global__ void cnt_kernel(const int* __restrict__ dst, float* __restrict__ cnt){
  int e=blockIdx.x*blockDim.x+threadIdx.x;
  if(e<NEDGE) atomicAdd(&cnt[dst[e]],1.f);
}

// ---------------------------------------------------------------------------
// Fused edge message kernel: m0 -> m1 -> m2 -> atomic scatter at dst.
// One wave per 32 edges (MT=2 tiles); LDS ping-pong bufA(9kt)/bufB(4kt).
// ---------------------------------------------------------------------------
__global__ __launch_bounds__(32) void edge_conv_kernel(
    const float* __restrict__ sN, const float* __restrict__ vN,
    const float* __restrict__ esE, const float* __restrict__ evE,
    const int* __restrict__ src, const int* __restrict__ dst,
    const __bf16* m0whP, const __bf16* m0wsP, const float* m0b, const __bf16* m0wvP,
    const __bf16* m1whP, const __bf16* m1wsP, const float* m1b, const __bf16* m1wvP,
    const __bf16* m2whP, const __bf16* m2wsP, const float* m2b, const __bf16* m2wvP,
    float* __restrict__ dhs, float* __restrict__ dhv){
  constexpr int AST=9*512, BST=4*512, VST=6*512;
  __shared__ __align__(32) __bf16 bufA[2*AST];
  __shared__ __align__(32) __bf16 bufB[2*BST];
  __shared__ __align__(32) __bf16 aV[2*VST];
  const int lane=threadIdx.x&31;
  const int e0=blockIdx.x*32;
  if(lane==0 && e0+32<NEDGE) __builtin_prefetch(esE+(size_t)(e0+32)*32, 0, 1);
  for(int i=lane;i<2*AST;i+=32) bufA[i]=(__bf16)0.f;
  for(int i=lane;i<2*BST;i+=32) bufB[i]=(__bf16)0.f;
  for(int i=lane;i<2*VST;i+=32) aV[i]=(__bf16)0.f;
  // stage A = [s_src(100) | edge_s(32) | s_dst(100)] for 32 edges (2 tiles)
  for(int i=lane;i<32*232;i+=32){
    int row=i/232, k=i-row*232, e=e0+row;
    int mt=row>>4, m=row&15;
    float x;
    if(k<100)       x=sN[(size_t)src[e]*100+k];
    else if(k<132)  x=esE[(size_t)e*32+(k-100)];
    else            x=sN[(size_t)dst[e]*100+(k-132)];
    stA(bufA+mt*AST,m,k,x);
  }
  // stage V = [v_src(16) | edge_v(1) | v_dst(16)] per coordinate
  for(int i=lane;i<32*99;i+=32){
    int row=i/99, t=i-row*99, vv=t/3, c=t-vv*3, e=e0+row;
    int mt=row>>4, m=row&15;
    float x;
    if(vv<16)       x=vN[((size_t)src[e]*16+vv)*3+c];
    else if(vv==16) x=evE[(size_t)e*3+c];
    else            x=vN[((size_t)dst[e]*16+(vv-17))*3+c];
    stA(aV+mt*VST+c*1024,m,vv,x);
  }
  LDSFENCE();
  v8f vO[3*2];   // [(c)*2+mt]
  gvp_tile<232,33,33,100,16,1,2>(bufA,aV,bufB,nullptr,vO,lane,
                                 m0whP,m0wsP,m0b,m0wvP,AST,VST,BST);
#pragma unroll
  for(int mt=0;mt<2;mt++){
    for(int i=lane;i<4*512;i+=32) bufA[mt*AST+i]=(__bf16)0.f;
    for(int i=lane;i<VST;i+=32)   aV[mt*VST+i]=(__bf16)0.f;
  }
  { int col=lane&15, mh=(lane>>4)*8;
#pragma unroll
    for(int mt=0;mt<2;mt++)
#pragma unroll
      for(int c=0;c<3;c++)
#pragma unroll
        for(int r=0;r<8;r++) stA(aV+mt*VST+c*1024,r+mh,col,vO[c*2+mt][r]); }
  LDSFENCE();
  gvp_tile<100,16,16,100,16,1,2>(bufB,aV,bufA,nullptr,vO,lane,
                                 m1whP,m1wsP,m1b,m1wvP,BST,VST,AST);
#pragma unroll
  for(int mt=0;mt<2;mt++)
    for(int i=lane;i<VST;i+=32) aV[mt*VST+i]=(__bf16)0.f;
  { int col=lane&15, mh=(lane>>4)*8;
#pragma unroll
    for(int mt=0;mt<2;mt++)
#pragma unroll
      for(int c=0;c<3;c++)
#pragma unroll
        for(int r=0;r<8;r++) stA(aV+mt*VST+c*1024,r+mh,col,vO[c*2+mt][r]); }
  LDSFENCE();
  v8f sO[7*2];   // [nt*2+mt]
  gvp_tile<100,16,16,100,16,0,2>(bufA,aV,nullptr,sO,vO,lane,
                                 m2whP,m2wsP,m2b,m2wvP,AST,VST,0);
  // mean-scatter (division by cnt happens in node_update)
#pragma unroll
  for(int nt=0;nt<7;nt++){
    int col=nt*16+(lane&15);
    if(col<100){
      int mh=(lane>>4)*8;
#pragma unroll
      for(int mt=0;mt<2;mt++)
#pragma unroll
        for(int r=0;r<8;r++){
          int e=e0+mt*16+r+mh;
          atomicAdd(&dhs[(size_t)dst[e]*100+col], sO[nt*2+mt][r]);
        }
    }
  }
  { int col=lane&15, mh=(lane>>4)*8;
#pragma unroll
    for(int mt=0;mt<2;mt++)
#pragma unroll
      for(int c=0;c<3;c++)
#pragma unroll
        for(int r=0;r<8;r++){
          int e=e0+mt*16+r+mh;
          atomicAdd(&dhv[((size_t)dst[e]*16+col)*3+c], vO[c*2+mt][r]);
        } }
}

// ---------------------------------------------------------------------------
// Node update: residual + LN(norm0) -> ff0 -> ff1 -> residual + LN(norm1).
// One wave per 16-node tile.
// ---------------------------------------------------------------------------
__global__ __launch_bounds__(32) void node_update_kernel(
    float* __restrict__ sN, float* __restrict__ vN,
    const float* __restrict__ dhs, const float* __restrict__ dhv,
    const float* __restrict__ cnt,
    const float* n0b,const float* n0w,const float* n1b,const float* n1w,
    const __bf16* f0whP,const float* f0b,const __bf16* f0wsP,const __bf16* f0wvP,
    const __bf16* f1whP,const float* f1b,const __bf16* f1wsP,const __bf16* f1wvP){
  __shared__ __align__(32) __bf16 aS[5*512];
  __shared__ __align__(32) __bf16 aS2[14*512];
  __shared__ __align__(32) __bf16 aV[6*512];
  __shared__ float sT[16][100];
  __shared__ float vT[16][16][3];
  const int lane=threadIdx.x, tile=blockIdx.x;
  if(lane<16){
    const int n=tile*16+lane;
    const float c=fmaxf(cnt[n],1.f);
    float mu=0.f;
    for(int k=0;k<100;k++){ float x=sN[(size_t)n*100+k]+dhs[(size_t)n*100+k]/c; sT[lane][k]=x; mu+=x; }
    mu*=0.01f;
    float var=0.f;
    for(int k=0;k<100;k++){ float d=sT[lane][k]-mu; var+=d*d; }
    var*=0.01f;
    float rs=rsqrtf(var+1e-5f);
    for(int k=0;k<100;k++) sT[lane][k]=(sT[lane][k]-mu)*rs*n0w[k]+n0b[k];
    float acc=0.f;
    for(int i=0;i<16;i++){
      float s2=0.f;
      for(int c3=0;c3<3;c3++){
        float x=vN[((size_t)n*16+i)*3+c3]+dhv[((size_t)n*16+i)*3+c3]/c;
        vT[lane][i][c3]=x; s2+=x*x;
      }
      acc+=fmaxf(s2,1e-8f);
    }
    float sc=rsqrtf(acc*(1.f/16.f));
    for(int i=0;i<16;i++) for(int c3=0;c3<3;c3++) vT[lane][i][c3]*=sc;
  }
  LDSFENCE();
  for(int i=lane;i<5*512;i+=32)  aS[i]=(__bf16)0.f;
  for(int i=lane;i<14*512;i+=32) aS2[i]=(__bf16)0.f;
  for(int i=lane;i<6*512;i+=32)  aV[i]=(__bf16)0.f;
  for(int i=lane;i<1600;i+=32){ int m=i/100,k=i-m*100; stA(aS,m,k,sT[m][k]); }
  for(int i=lane;i<768;i+=32){ int m=i/48,t=i-m*48,vv=t/3,c=t-vv*3; stA(aV+c*1024,m,vv,vT[m][vv][c]); }
  LDSFENCE();
  v8f vO0[6];
  gvp_tile<100,16,32,400,32,1,1>(aS,aV,aS2,nullptr,vO0,lane,f0whP,f0wsP,f0b,f0wvP);
  for(int i=lane;i<6*512;i+=32) aV[i]=(__bf16)0.f;
#pragma unroll
  for(int c=0;c<3;c++)
#pragma unroll
    for(int nt=0;nt<2;nt++){
      int col=nt*16+(lane&15), mh=(lane>>4)*8;
#pragma unroll
      for(int r=0;r<8;r++) stA(aV+c*1024,r+mh,col,vO0[c*2+nt][r]);
    }
  LDSFENCE();
  v8f sO[7]; v8f vO1[3];
  gvp_tile<400,32,32,100,16,0,1>(aS2,aV,nullptr,sO,vO1,lane,f1whP,f1wsP,f1b,f1wvP);
#pragma unroll
  for(int nt=0;nt<7;nt++){
    int col=nt*16+(lane&15);
    if(col<100){
      int mh=(lane>>4)*8;
#pragma unroll
      for(int r=0;r<8;r++) sT[r+mh][col]+=sO[nt][r];
    }
  }
  { int col=lane&15, mh=(lane>>4)*8;
#pragma unroll
    for(int c=0;c<3;c++)
#pragma unroll
      for(int r=0;r<8;r++) vT[r+mh][col][c]+=vO1[c][r]; }
  LDSFENCE();
  if(lane<16){
    const int n=tile*16+lane;
    float mu=0.f;
    for(int k=0;k<100;k++) mu+=sT[lane][k];
    mu*=0.01f;
    float var=0.f;
    for(int k=0;k<100;k++){ float d=sT[lane][k]-mu; var+=d*d; }
    var*=0.01f;
    float rs=rsqrtf(var+1e-5f);
    for(int k=0;k<100;k++) sN[(size_t)n*100+k]=(sT[lane][k]-mu)*rs*n1w[k]+n1b[k];
    float acc=0.f;
    for(int i=0;i<16;i++){
      float s2=0.f;
      for(int c3=0;c3<3;c3++){ float x=vT[lane][i][c3]; s2+=x*x; }
      acc+=fmaxf(s2,1e-8f);
    }
    float sc=rsqrtf(acc*(1.f/16.f));
    for(int i=0;i<16;i++) for(int c3=0;c3<3;c3++)
      vN[((size_t)n*16+i)*3+c3]=vT[lane][i][c3]*sc;
  }
}

// ---------------------------------------------------------------------------
// Policy & value node heads: LN + GVP(100,16 -> 20,0). Policy path writes the
// masked [16 x 5120] matrix straight into bf16 A-fragment layout for the
// policy MLP; value path scatter-sums per graph.
// ---------------------------------------------------------------------------
__global__ __launch_bounds__(32) void heads_kernel(
    const float* __restrict__ sN, const float* __restrict__ vN,
    const float* __restrict__ avail,
    const float* plnb,const float* plnw,const __bf16* pwhP,const float* pwsb,const __bf16* pwsP,
    const float* vlnb,const float* vlnw,const __bf16* vwhP,const float* vwsb,const __bf16* vwsP,
    __bf16* __restrict__ polA, float* __restrict__ vsum){
  __shared__ __align__(32) __bf16 aS[4*512];
  __shared__ __align__(32) __bf16 aV[6*512];
  __shared__ float sT[16][100];
  __shared__ float vT[16][16][3];
  const int lane=threadIdx.x, tile=blockIdx.x;
  for(int head=0; head<2; ++head){
    const float* lb = head? vlnb:plnb; const float* lw = head? vlnw:plnw;
    const __bf16* whP = head? vwhP:pwhP; const __bf16* wsP = head? vwsP:pwsP;
    const float* wsb = head? vwsb:pwsb;
    if(lane<16){
      const int n=tile*16+lane;
      float mu=0.f;
      for(int k=0;k<100;k++){ float x=sN[(size_t)n*100+k]; sT[lane][k]=x; mu+=x; }
      mu*=0.01f;
      float var=0.f;
      for(int k=0;k<100;k++){ float d=sT[lane][k]-mu; var+=d*d; }
      var*=0.01f;
      float rs=rsqrtf(var+1e-5f);
      for(int k=0;k<100;k++) sT[lane][k]=(sT[lane][k]-mu)*rs*lw[k]+lb[k];
      float acc=0.f;
      for(int i=0;i<16;i++){
        float s2=0.f;
        for(int c3=0;c3<3;c3++){ float x=vN[((size_t)n*16+i)*3+c3]; vT[lane][i][c3]=x; s2+=x*x; }
        acc+=fmaxf(s2,1e-8f);
      }
      float sc=rsqrtf(acc*(1.f/16.f));
      for(int i=0;i<16;i++) for(int c3=0;c3<3;c3++) vT[lane][i][c3]*=sc;
    }
    LDSFENCE();
    for(int i=lane;i<4*512;i+=32) aS[i]=(__bf16)0.f;
    for(int i=lane;i<6*512;i+=32) aV[i]=(__bf16)0.f;
    for(int i=lane;i<1600;i+=32){ int m=i/100,k=i-m*100; stA(aS,m,k,sT[m][k]); }
    for(int i=lane;i<768;i+=32){ int m=i/48,t=i-m*48,vv=t/3,c=t-vv*3; stA(aV+c*1024,m,vv,vT[m][vv][c]); }
    LDSFENCE();
    v8f sO[2];
    gvp_tile<100,16,16,20,0,0,1>(aS,aV,nullptr,sO,nullptr,lane,whP,wsP,wsb,nullptr);
#pragma unroll
    for(int nt=0;nt<2;nt++){
      int col=nt*16+(lane&15);
      if(col<20){
        int mh=(lane>>4)*8;
#pragma unroll
        for(int r=0;r<8;r++){
          int n=tile*16+r+mh;
          int b=n>>8, l=n&255;
          float val=sO[nt][r];
          if(head==0){
            val*=avail[b*256+l];
            int k=l*20+col;
            polA[aidx(b,k)]=(__bf16)val;   // [16 x 5120] in A-fragment layout
          }else{
            atomicAdd(&vsum[b*20+col], val);
          }
        }
      }
    }
  }
}

// ---------------------------------------------------------------------------
// Policy MLP: 16x5120 @ 5120x5120 (x2). One wave per 16-col tile, 160 K-tiles.
// A read as bf16 fragments; W streamed once from HBM as f32, cvt to bf16.
// ---------------------------------------------------------------------------
__global__ __launch_bounds__(64) void dense16_kernel(
    const __bf16* __restrict__ Ain, const float* __restrict__ W,
    const float* __restrict__ bias, __bf16* __restrict__ outB, float* __restrict__ outF){
  const int wave=threadIdx.x>>5, lane=threadIdx.x&31;
  const int nt=blockIdx.x*2+wave;
  const float* Wa=(const float*)__builtin_assume_aligned(W,16);
  const int n=nt*16+(lane&15);
  const int half=lane>>4;
  v8f acc=vzero();
  for(int kt=0;kt<160;kt++){
    v16bf a=ldA(Ain,kt,lane);
    const float* p=Wa + (size_t)n*5120 + kt*32 + half*8;
    float4 q0=*(const float4*)(p);
    float4 q1=*(const float4*)(p+4);
    float4 q2=*(const float4*)(p+16);
    float4 q3=*(const float4*)(p+20);
    v16bf b;
    b[0]=(__bf16)q0.x; b[1]=(__bf16)q0.y; b[2]=(__bf16)q0.z; b[3]=(__bf16)q0.w;
    b[4]=(__bf16)q1.x; b[5]=(__bf16)q1.y; b[6]=(__bf16)q1.z; b[7]=(__bf16)q1.w;
    b[8]=(__bf16)q2.x; b[9]=(__bf16)q2.y; b[10]=(__bf16)q2.z; b[11]=(__bf16)q2.w;
    b[12]=(__bf16)q3.x; b[13]=(__bf16)q3.y; b[14]=(__bf16)q3.z; b[15]=(__bf16)q3.w;
    acc=wmma_bf16(a,b,acc);
  }
  const int col=nt*16+(lane&15);
  const float bb=bias[col];
#pragma unroll
  for(int r=0;r<8;r++){
    float x=acc[r]+bb;
    int m=r+8*half;
    if(outF){
      outF[(size_t)m*5120+col]=x;
    }else{
      x=fmaxf(x,0.f);
      outB[aidx(m,col)]=(__bf16)x;
    }
  }
}

__global__ void value_out_kernel(const float* __restrict__ vsum,
    const float* w1,const float* b1,const float* w2,const float* b2,
    float* __restrict__ out){
  __shared__ float hid[16*10];
  const int t=threadIdx.x;
  if(t<160){
    int g=t/10, j=t%10;
    float x=b1[j];
    for(int k=0;k<20;k++) x+=vsum[g*20+k]*w1[j*20+k];
    hid[g*10+j]=(x>0.f)?x:0.01f*x;
  }
  __syncthreads();
  for(int i=t;i<16*601;i+=blockDim.x){
    int g=i/601, o=i%601;
    float x=b2[o];
    for(int j=0;j<10;j++) x+=hid[g*10+j]*w2[o*10+j];
    out[(size_t)g*601+o]=x;
  }
}

// ---------------------------------------------------------------------------
// Host-side orchestration. Param leaves assumed flattened depth-first with
// sorted dict keys (jax tree order); lists in order.
// ---------------------------------------------------------------------------
extern "C" void kernel_launch(void* const* d_in, const int* in_sizes, int n_in,
                              void* d_out, int out_size, void* d_ws, size_t ws_size,
                              hipStream_t stream){
  (void)in_sizes; (void)n_in; (void)out_size; (void)ws_size;
  auto F=[&](int i){ return (const float*)d_in[i]; };
  const float* node_s=F(0); const float* node_v=F(1);
  const float* edge_s=F(2); const float* edge_v=F(3);
  const float* avail =F(4);
  const float* ee_wh=F(5); const float* ee_wsb=F(6); const float* ee_wsw=F(7); const float* ee_wv=F(8);
  const float* ee_lnb=F(9); const float* ee_lnw=F(10);
  const int LBASE=11;                                  // 3 layers x 24 leaves
  const float* ne_wh=F(83); const float* ne_wsb=F(84); const float* ne_wsw=F(85); const float* ne_wv=F(86);
  const float* ne_lnb=F(87); const float* ne_lnw=F(88);
  const float* pg_wh=F(89); const float* pg_wsb=F(90); const float* pg_wsw=F(91);
  const float* p_lnb=F(92); const float* p_lnw=F(93);
  const float* po_b1=F(94); const float* po_b2=F(95); const float* po_w1=F(96); const float* po_w2=F(97);
  const float* vg_wh=F(98); const float* vg_wsb=F(99); const float* vg_wsw=F(100);
  const float* v_lnb=F(101); const float* v_lnw=F(102);
  const float* vo_b1=F(103); const float* vo_b2=F(104); const float* vo_w1=F(105); const float* vo_w2=F(106);
  const int* ei=(const int*)d_in[107];
  const int* srcp=ei; const int* dstp=ei+NEDGE;

  char* ws=(char*)d_ws; size_t off=0;
  auto alloc=[&](size_t b)->void*{ void* p=ws+off; off+=(b+255)&~(size_t)255; return p; };
  float*  sN  =(float*)alloc((size_t)NNODE*100*4);
  float*  vN  =(float*)alloc((size_t)NNODE*48*4);
  float*  esE =(float*)alloc((size_t)NEDGE*32*4);
  float*  evE =(float*)alloc((size_t)NEDGE*3*4);
  float*  dhs =(float*)alloc((size_t)NNODE*100*4);
  float*  dhv =(float*)alloc((size_t)NNODE*48*4);
  float*  cnt =(float*)alloc((size_t)NNODE*4);
  __bf16* polA=(__bf16*)alloc((size_t)160*512*2);
  __bf16* hidA=(__bf16*)alloc((size_t)160*512*2);
  float*  vsum=(float*)alloc((size_t)16*20*4);

  auto packW=[&](const float* W,int Nn,int Kk)->const __bf16*{
    int KT=(Kk+31)/32, NT=(Nn+15)/16;
    size_t el=(size_t)KT*NT*512;
    __bf16* p=(__bf16*)alloc(el*2);
    int blocks=(int)((el+255)/256);
    pack_b_kernel<<<blocks,256,0,stream>>>(W,p,Nn,Kk);
    return p;
  };

  hipMemsetAsync(cnt,0,(size_t)NNODE*4,stream);
  hipMemsetAsync(vsum,0,(size_t)16*20*4,stream);
  embed_node_kernel<<<NNODE/256,256,0,stream>>>(node_s,node_v,ne_lnb,ne_lnw,ne_wh,ne_wsb,ne_wsw,ne_wv,sN,vN);
  embed_edge_kernel<<<NEDGE/256,256,0,stream>>>(edge_s,edge_v,ee_lnb,ee_lnw,ee_wh,ee_wsb,ee_wsw,ee_wv,esE,evE);
  cnt_kernel<<<NEDGE/256,256,0,stream>>>(dstp,cnt);

  const __bf16* pg_whP=packW(pg_wh,16,16);
  const __bf16* pg_wsP=packW(pg_wsw,20,116);
  const __bf16* vg_whP=packW(vg_wh,16,16);
  const __bf16* vg_wsP=packW(vg_wsw,20,116);

  for(int l=0;l<3;l++){
    int b=LBASE+24*l;
    const __bf16* m0whP=packW(F(b+0),33,33);   const float* m0b=F(b+1);
    const __bf16* m0wsP=packW(F(b+2),100,265); const __bf16* m0wvP=packW(F(b+3),16,33);
    const __bf16* m1whP=packW(F(b+4),16,16);   const float* m1b=F(b+5);
    const __bf16* m1wsP=packW(F(b+6),100,116); const __bf16* m1wvP=packW(F(b+7),16,16);
    const __bf16* m2whP=packW(F(b+8),16,16);   const float* m2b=F(b+9);
    const __bf16* m2wsP=packW(F(b+10),100,116);const __bf16* m2wvP=packW(F(b+11),16,16);
    const __bf16* f0whP=packW(F(b+12),32,16);  const float* f0b=F(b+13);
    const __bf16* f0wsP=packW(F(b+14),400,132);const __bf16* f0wvP=packW(F(b+15),32,32);
    const __bf16* f1whP=packW(F(b+16),32,32);  const float* f1b=F(b+17);
    const __bf16* f1wsP=packW(F(b+18),100,432);const __bf16* f1wvP=packW(F(b+19),16,32);
    const float* n0b=F(b+20); const float* n0w=F(b+21);
    const float* n1b=F(b+22); const float* n1w=F(b+23);

    hipMemsetAsync(dhs,0,(size_t)NNODE*100*4,stream);
    hipMemsetAsync(dhv,0,(size_t)NNODE*48*4,stream);
    edge_conv_kernel<<<NEDGE/32,32,0,stream>>>(sN,vN,esE,evE,srcp,dstp,
        m0whP,m0wsP,m0b,m0wvP, m1whP,m1wsP,m1b,m1wvP, m2whP,m2wsP,m2b,m2wvP,
        dhs,dhv);
    node_update_kernel<<<NNODE/16,32,0,stream>>>(sN,vN,dhs,dhv,cnt,
        n0b,n0w,n1b,n1w, f0whP,f0b,f0wsP,f0wvP, f1whP,f1b,f1wsP,f1wvP);
  }

  heads_kernel<<<NNODE/16,32,0,stream>>>(sN,vN,avail,
      p_lnb,p_lnw,pg_whP,pg_wsb,pg_wsP,
      v_lnb,v_lnw,vg_whP,vg_wsb,vg_wsP, polA,vsum);

  float* outPol=(float*)d_out;
  float* outVal=outPol+(size_t)16*5120;
  dense16_kernel<<<160,64,0,stream>>>(polA,po_w1,po_b1,hidA,nullptr);
  dense16_kernel<<<160,64,0,stream>>>(hidA,po_w2,po_b2,nullptr,outPol);
  value_out_kernel<<<1,256,0,stream>>>(vsum,vo_w1,vo_b1,vo_w2,vo_b2,outVal);
}
